// ChannelAttGatedGRUCell_13460427505843
// MI455X (gfx1250) — matrix operand order repaired and checked by reading, not compiled, not run on hardware
//
#include <hip/hip_runtime.h>

typedef __attribute__((ext_vector_type(16))) _Float16 v16h;
typedef __attribute__((ext_vector_type(8)))  float    v8f;

#define BC 32768  // B*C = 64*512

// ---------------------------------------------------------------------------
// Weight convert + transpose: W (K x N, f32, row-major) -> WT (N x K, f16)
// Grid covers exactly N*K elements (all sizes multiples of 256).
// ---------------------------------------------------------------------------
__global__ __launch_bounds__(256) void transpose_w_kernel(const float* __restrict__ W,
                                                          _Float16* __restrict__ WT,
                                                          int K, int N) {
  const long long idx = (long long)blockIdx.x * 256 + threadIdx.x;
  const int n = (int)(idx / K);
  const int k = (int)(idx % K);
  WT[idx] = (_Float16)W[(size_t)k * N + n];
}

// ---------------------------------------------------------------------------
// Pack xh = concat([x, h], axis=-1) as f16:  (BC, 2048)
// ---------------------------------------------------------------------------
__global__ __launch_bounds__(256) void pack_xh_kernel(const float* __restrict__ x,
                                                      const float* __restrict__ h,
                                                      _Float16* __restrict__ xh) {
  const long long idx = (long long)blockIdx.x * 256 + threadIdx.x;
  const long long m = idx >> 11;
  const int c = (int)(idx & 2047);
  const float v = (c < 1024) ? x[m * 1024 + c] : h[m * 1024 + (c - 1024)];
  xh[idx] = (_Float16)v;
}

// ---------------------------------------------------------------------------
// Generic WMMA GEMM:  D = epilogue(A @ BT^T)    (A: MxK f16, BT: NxK f16)
// Block tile 128x128, BK=32, 8 waves, per-wave 32x64 (2x4 wmma accumulators).
// Double-buffered LDS: one barrier per K-step.
// EPI: 0 = f16 row-major store + bias[n]
//      1 = f16 TRANSPOSED store (D[n][m]) + bias[n]      (for V^T)
//      2 = f32 store * scale                             (scores)
//      3 = f32 store sigmoid                             (update gate)
//      4 = f32 store tanh                                (candidate n)
//      5 = f16 store of extra[m][n] * sigmoid(acc)       (h * reset -> xh)
// ---------------------------------------------------------------------------
template<int EPI>
__global__ __launch_bounds__(256) void wmma_gemm(
    const _Float16* __restrict__ A, int lda, long long sA,
    const _Float16* __restrict__ BT, int ldb, long long sB,
    void* __restrict__ Dv, int ldd, long long sD,
    const float* __restrict__ bias,
    const float* __restrict__ extra, int lde, long long sE,
    float scale, int M, int N, int K) {
  (void)M; (void)N;
  __shared__ __align__(16) _Float16 As[2][128][40];  // [buf][m][k], pad 40
  __shared__ __align__(16) _Float16 Bs[2][128][40];  // [buf][n][k], pad 40

  const int tid  = threadIdx.x;
  const int lane = tid & 31;
  const int wave = tid >> 5;
  const int wm   = wave & 3;   // 4 waves along M  (32 rows each)
  const int wn   = wave >> 2;  // 2 waves along N  (64 cols each)
  const int l16  = lane >> 4;  // 0/1 half-wave
  const int lmod = lane & 15;

  const int m0 = blockIdx.x * 128;
  const int n0 = blockIdx.y * 128;
  A  += (size_t)blockIdx.z * (size_t)sA;
  BT += (size_t)blockIdx.z * (size_t)sB;

  // staging: 256 threads, each moves 16 contiguous halves of A and of BT
  const int sr = tid >> 1;          // 0..127 tile row
  const int sc = (tid & 1) * 16;    // 0 or 16 within BK=32
  const _Float16* gA = A  + (size_t)(m0 + sr) * lda + sc;
  const _Float16* gB = BT + (size_t)(n0 + sr) * ldb + sc;

  v8f acc[2][4] = {};

  // prologue: stage K-slice 0 into buffer 0
  {
    const uint4* ga = (const uint4*)gA;
    const uint4* gb = (const uint4*)gB;
    const uint4 a0 = ga[0], a1 = ga[1];
    const uint4 b0 = gb[0], b1 = gb[1];
    *(uint4*)&As[0][sr][sc]     = a0;
    *(uint4*)&As[0][sr][sc + 8] = a1;
    *(uint4*)&Bs[0][sr][sc]     = b0;
    *(uint4*)&Bs[0][sr][sc + 8] = b1;
  }
  __syncthreads();

  int cur = 0;
  for (int k0 = 0; k0 < K; k0 += 32, cur ^= 1) {
    const bool has_next = (k0 + 32) < K;
    uint4 a0{}, a1{}, b0{}, b1{};
    if (has_next) {
      const uint4* ga = (const uint4*)(gA + k0 + 32);
      const uint4* gb = (const uint4*)(gB + k0 + 32);
      a0 = ga[0]; a1 = ga[1]; b0 = gb[0]; b1 = gb[1];
      if ((k0 + 64) < K) {  // keep L2->WGP pipe primed (global_prefetch_b8)
        __builtin_prefetch(gA + k0 + 64, 0, 1);
        __builtin_prefetch(gB + k0 + 64, 0, 1);
      }
    }

    // A fragment: lane holds row m=lmod; k = {l16*8..+7} and {16+l16*8..+7}
    v16h afrag[2], bfrag[4];
#pragma unroll
    for (int mt = 0; mt < 2; ++mt) {
      const _Float16* ap = &As[cur][wm * 32 + mt * 16 + lmod][l16 * 8];
      union { uint4 u[2]; v16h v; } t;
      t.u[0] = *(const uint4*)ap;
      t.u[1] = *(const uint4*)(ap + 16);
      afrag[mt] = t.v;
    }
    // B fragment: lane holds col n=lmod; k = {l16*8..+7} and {16+l16*8..+7}
#pragma unroll
    for (int nt = 0; nt < 4; ++nt) {
      const _Float16* bp = &Bs[cur][wn * 64 + nt * 16 + lmod][l16 * 8];
      union { uint4 u[2]; v16h v; } t;
      t.u[0] = *(const uint4*)bp;
      t.u[1] = *(const uint4*)(bp + 16);
      bfrag[nt] = t.v;
    }
#pragma unroll
    for (int mt = 0; mt < 2; ++mt)
#pragma unroll
      for (int nt = 0; nt < 4; ++nt)
        acc[mt][nt] = __builtin_amdgcn_wmma_f32_16x16x32_f16(
            false, afrag[mt], false, bfrag[nt], (short)0, acc[mt][nt],
            false, false);

    if (has_next) {  // fill the other buffer; nobody is reading it this step
      const int nxt = cur ^ 1;
      *(uint4*)&As[nxt][sr][sc]     = a0;
      *(uint4*)&As[nxt][sr][sc + 8] = a1;
      *(uint4*)&Bs[nxt][sr][sc]     = b0;
      *(uint4*)&Bs[nxt][sr][sc + 8] = b1;
    }
    __syncthreads();  // single barrier per K-step
  }

  // Epilogue. C/D layout: lane l, vgpr v -> n = lmod, m = l16*8 + v (per tile)
#pragma unroll
  for (int mt = 0; mt < 2; ++mt) {
#pragma unroll
    for (int nt = 0; nt < 4; ++nt) {
      const int mg = m0 + wm * 32 + mt * 16 + l16 * 8;
      const int ng = n0 + wn * 64 + nt * 16 + lmod;
      if constexpr (EPI == 0) {
        _Float16* D = (_Float16*)Dv + (size_t)blockIdx.z * (size_t)sD;
        const float bb = bias[ng];
#pragma unroll
        for (int v = 0; v < 8; ++v)
          D[(size_t)(mg + v) * ldd + ng] = (_Float16)(acc[mt][nt][v] + bb);
      } else if constexpr (EPI == 1) {
        _Float16* D = (_Float16*)Dv + (size_t)blockIdx.z * (size_t)sD;
        const float bb = bias[ng];
        union { uint4 u; _Float16 hh[8]; } pk;
#pragma unroll
        for (int v = 0; v < 8; ++v) pk.hh[v] = (_Float16)(acc[mt][nt][v] + bb);
        *(uint4*)&D[(size_t)ng * ldd + mg] = pk.u;  // contiguous transposed store
      } else if constexpr (EPI == 2) {
        float* D = (float*)Dv + (size_t)blockIdx.z * (size_t)sD;
#pragma unroll
        for (int v = 0; v < 8; ++v)
          D[(size_t)(mg + v) * ldd + ng] = acc[mt][nt][v] * scale;
      } else if constexpr (EPI == 3) {
        float* D = (float*)Dv + (size_t)blockIdx.z * (size_t)sD;
#pragma unroll
        for (int v = 0; v < 8; ++v)
          D[(size_t)(mg + v) * ldd + ng] = 1.f / (1.f + __expf(-acc[mt][nt][v]));
      } else if constexpr (EPI == 4) {
        float* D = (float*)Dv + (size_t)blockIdx.z * (size_t)sD;
#pragma unroll
        for (int v = 0; v < 8; ++v)
          D[(size_t)(mg + v) * ldd + ng] = tanhf(acc[mt][nt][v]);
      } else {  // EPI == 5: xh_h_half[m][n] = f16( h[m][n] * sigmoid(acc) )
        _Float16* D = (_Float16*)Dv + (size_t)blockIdx.z * (size_t)sD;
        const float* E = extra + (size_t)blockIdx.z * (size_t)sE;
#pragma unroll
        for (int v = 0; v < 8; ++v) {
          const float sg = 1.f / (1.f + __expf(-acc[mt][nt][v]));
          D[(size_t)(mg + v) * ldd + ng] =
              (_Float16)(E[(size_t)(mg + v) * lde + ng] * sg);
        }
      }
    }
  }
}

// ---------------------------------------------------------------------------
// Row softmax over C=512, f32 in -> f16 probs out. One block per row.
// ---------------------------------------------------------------------------
__global__ __launch_bounds__(256) void softmax_kernel(const float* __restrict__ S,
                                                      _Float16* __restrict__ P) {
  const int tid = threadIdx.x;
  const size_t base = (size_t)blockIdx.x * 512;
  const float v0 = S[base + tid];
  const float v1 = S[base + tid + 256];
  float m = fmaxf(v0, v1);
  __shared__ float red[8];
#pragma unroll
  for (int off = 16; off > 0; off >>= 1) m = fmaxf(m, __shfl_xor(m, off, 32));
  if ((tid & 31) == 0) red[tid >> 5] = m;
  __syncthreads();
  float mall = red[0];
#pragma unroll
  for (int w = 1; w < 8; ++w) mall = fmaxf(mall, red[w]);
  __syncthreads();
  const float e0 = __expf(v0 - mall);
  const float e1 = __expf(v1 - mall);
  float s = e0 + e1;
#pragma unroll
  for (int off = 16; off > 0; off >>= 1) s += __shfl_xor(s, off, 32);
  if ((tid & 31) == 0) red[tid >> 5] = s;
  __syncthreads();
  float tot = 0.f;
#pragma unroll
  for (int w = 0; w < 8; ++w) tot += red[w];
  const float inv = 1.f / tot;
  P[base + tid]       = (_Float16)(e0 * inv);
  P[base + tid + 256] = (_Float16)(e1 * inv);
}

// ---------------------------------------------------------------------------
// Final gating + LayerNorm over D_H=1024. One block per (b,c) row.
// new = (1-u)*h + u*tanh_n ; out = (new-mu)*rsqrt(var+eps)*gamma+beta
// ---------------------------------------------------------------------------
__global__ __launch_bounds__(256) void gate_ln_kernel(
    const float* __restrict__ h, const float* __restrict__ u,
    const float* __restrict__ tn, const float* __restrict__ gamma,
    const float* __restrict__ beta, float* __restrict__ out) {
  const int tid = threadIdx.x;
  const size_t base = (size_t)blockIdx.x * 1024;
  float vals[4];
  float sum = 0.f, sq = 0.f;
#pragma unroll
  for (int i = 0; i < 4; ++i) {
    const int c = tid + i * 256;
    const float uu = u[base + c];
    const float v = (1.f - uu) * h[base + c] + uu * tn[base + c];
    vals[i] = v; sum += v; sq += v * v;
  }
  __shared__ float rs[8], rq[8];
#pragma unroll
  for (int off = 16; off > 0; off >>= 1) {
    sum += __shfl_xor(sum, off, 32);
    sq  += __shfl_xor(sq,  off, 32);
  }
  if ((tid & 31) == 0) { rs[tid >> 5] = sum; rq[tid >> 5] = sq; }
  __syncthreads();
  sum = 0.f; sq = 0.f;
#pragma unroll
  for (int w = 0; w < 8; ++w) { sum += rs[w]; sq += rq[w]; }
  const float mean = sum * (1.f / 1024.f);
  const float var  = sq * (1.f / 1024.f) - mean * mean;
  const float rstd = rsqrtf(var + 1e-5f);
#pragma unroll
  for (int i = 0; i < 4; ++i) {
    const int c = tid + i * 256;
    out[base + c] = (vals[i] - mean) * rstd * gamma[c] + beta[c];
  }
}

// ---------------------------------------------------------------------------
// Host side
// ---------------------------------------------------------------------------
static void* carve(char*& p, size_t bytes) {
  void* r = (void*)p;
  p += (bytes + 255) & ~(size_t)255;
  return r;
}

extern "C" void kernel_launch(void* const* d_in, const int* in_sizes, int n_in,
                              void* d_out, int out_size, void* d_ws, size_t ws_size,
                              hipStream_t stream) {
  (void)in_sizes; (void)n_in; (void)out_size; (void)ws_size;

  const float* x = (const float*)d_in[0];
  const float* h = (const float*)d_in[1];
  // gate order in inputs: r, z, n
  const float* Wq[3] = {(const float*)d_in[2], (const float*)d_in[8],  (const float*)d_in[14]};
  const float* bq[3] = {(const float*)d_in[3], (const float*)d_in[9],  (const float*)d_in[15]};
  const float* Wk[3] = {(const float*)d_in[4], (const float*)d_in[10], (const float*)d_in[16]};
  const float* bk[3] = {(const float*)d_in[5], (const float*)d_in[11], (const float*)d_in[17]};
  const float* Wv[3] = {(const float*)d_in[6], (const float*)d_in[12], (const float*)d_in[18]};
  const float* bv[3] = {(const float*)d_in[7], (const float*)d_in[13], (const float*)d_in[19]};
  const float* gamma = (const float*)d_in[20];
  const float* beta  = (const float*)d_in[21];
  float* out = (float*)d_out;

  char* p = (char*)d_ws;
  _Float16* xh = (_Float16*)carve(p, (size_t)BC * 2048 * 2);     // 128 MB
  _Float16 *WqT[3], *WkT[3], *WvT[3];
  for (int g = 0; g < 3; ++g) {
    WqT[g] = (_Float16*)carve(p, (size_t)512  * 2048 * 2);
    WkT[g] = (_Float16*)carve(p, (size_t)512  * 2048 * 2);
    WvT[g] = (_Float16*)carve(p, (size_t)1024 * 2048 * 2);
  }
  _Float16* Qb = (_Float16*)carve(p, (size_t)BC * 512 * 2);      // 32 MB
  _Float16* Kb = (_Float16*)carve(p, (size_t)BC * 512 * 2);      // 32 MB
  _Float16* Vt = (_Float16*)carve(p, (size_t)1024 * BC * 2);     // 64 MB (transposed)
  float*    Sc = (float*)   carve(p, (size_t)64 * 512 * 512 * 4);// 64 MB scores
  _Float16* Pr = (_Float16*)carve(p, (size_t)64 * 512 * 512 * 2);// 32 MB probs
  float*    Ub = (float*)   carve(p, (size_t)BC * 1024 * 4);     // 128 MB update
  float*    Nb = (float*)   carve(p, (size_t)BC * 1024 * 4);     // 128 MB tanh(n)

  const dim3 blk(256);
  const float qk_scale = 0.04419417382415922f;  // 1/sqrt(512)

  // --- one-time per call: weight convert+transpose, pack xh ---
  for (int g = 0; g < 3; ++g) {
    transpose_w_kernel<<<4096, blk, 0, stream>>>(Wq[g], WqT[g], 2048, 512);
    transpose_w_kernel<<<4096, blk, 0, stream>>>(Wk[g], WkT[g], 2048, 512);
    transpose_w_kernel<<<8192, blk, 0, stream>>>(Wv[g], WvT[g], 2048, 1024);
  }
  pack_xh_kernel<<<(BC * 2048) / 256, blk, 0, stream>>>(x, h, xh);

  // one gate's projections + scores + softmax
  auto gate_front = [&](int g) {
    wmma_gemm<0><<<dim3(256, 4, 1), blk, 0, stream>>>(
        xh, 2048, 0, WqT[g], 2048, 0, Qb, 512, 0, bq[g],
        nullptr, 0, 0, 1.f, BC, 512, 2048);
    wmma_gemm<0><<<dim3(256, 4, 1), blk, 0, stream>>>(
        xh, 2048, 0, WkT[g], 2048, 0, Kb, 512, 0, bk[g],
        nullptr, 0, 0, 1.f, BC, 512, 2048);
    wmma_gemm<1><<<dim3(256, 8, 1), blk, 0, stream>>>(
        xh, 2048, 0, WvT[g], 2048, 0, Vt, BC, 0, bv[g],
        nullptr, 0, 0, 1.f, BC, 1024, 2048);
    wmma_gemm<2><<<dim3(4, 4, 64), blk, 0, stream>>>(
        Qb, 512, 262144, Kb, 512, 262144, Sc, 512, 262144, nullptr,
        nullptr, 0, 0, qk_scale, 512, 512, 512);
    softmax_kernel<<<BC, blk, 0, stream>>>(Sc, Pr);
  };

  // ---- z gate (idx 1): update = sigmoid(attn) ----
  gate_front(1);
  wmma_gemm<3><<<dim3(4, 8, 64), blk, 0, stream>>>(
      Pr, 512, 262144, Vt, BC, 512, Ub, 1024, 524288, nullptr,
      nullptr, 0, 0, 1.f, 512, 1024, 512);

  // ---- r gate (idx 0): xh[:,1024:] = f16(h * sigmoid(attn)) ----
  gate_front(0);
  wmma_gemm<5><<<dim3(4, 8, 64), blk, 0, stream>>>(
      Pr, 512, 262144, Vt, BC, 512, xh + 1024, 2048, 1048576, nullptr,
      h, 1024, 524288, 1.f, 512, 1024, 512);

  // ---- n gate (idx 2): reads updated xh; tanh fused ----
  gate_front(2);
  wmma_gemm<4><<<dim3(4, 8, 64), blk, 0, stream>>>(
      Pr, 512, 262144, Vt, BC, 512, Nb, 1024, 524288, nullptr,
      nullptr, 0, 0, 1.f, 512, 1024, 512);

  // ---- gating + LayerNorm ----
  gate_ln_kernel<<<BC, blk, 0, stream>>>(h, Ub, Nb, gamma, beta, out);
}
